// Conv2d_NN_Attn_Spatial_44976897523816
// MI455X (gfx1250) — matrix operand order repaired and checked by reading, not compiled, not run on hardware
//
#include <hip/hip_runtime.h>
#include <cmath>

// ---------------------------------------------------------------------------
// Problem constants (from reference)
// ---------------------------------------------------------------------------
#define NB     32      // batch
#define CIN    64
#define COUT   64
#define HW     64      // H = W = 64
#define NSP    1024    // 32*32 spatial positions after unshuffle
#define MS     256     // SAMPLES^2
#define C1     264     // (64+2)*4
#define CP     288     // C1 padded: multiple of 32 (WMMA K) and 16 (M/N)
#define TOPK   3
#define CPW    66      // C_OUT + 2

typedef _Float16 half16 __attribute__((ext_vector_type(16)));
typedef _Float16 half8  __attribute__((ext_vector_type(8)));
typedef float    f32x8  __attribute__((ext_vector_type(8)));

// ---------------------------------------------------------------------------
// K0a: sample indices. flat_idx[m] = xi[m/16]*32 + xi[m%16]; inv_idx = inverse
// ---------------------------------------------------------------------------
__global__ void k_setup_idx(int* flat_idx, int* inv_idx) {
  __shared__ int s_xi[16];
  int t = threadIdx.x;                       // blockDim = 1024
  if (t < 16) s_xi[t] = (int)rintf((float)t * 31.0f / 15.0f);
  if (t < NSP) inv_idx[t] = -1;
  __syncthreads();
  if (t < MS) {
    int f = s_xi[t >> 4] * 32 + s_xi[t & 15];
    flat_idx[t] = f;
    inv_idx[f]  = t;
  }
}

// ---------------------------------------------------------------------------
// K0b: convert Wq / Wk to f16, zero-padded to CP x CP (row-major [o][c])
// ---------------------------------------------------------------------------
__global__ void k_cvt_w(const float* __restrict__ Wq, const float* __restrict__ Wk,
                        _Float16* __restrict__ Wqh, _Float16* __restrict__ Wkh) {
  int t = blockIdx.x * blockDim.x + threadIdx.x;     // 2*CP*CP threads
  int which = t / (CP * CP);
  int r = t % (CP * CP);
  int o = r / CP, c = r % CP;
  float v = 0.f;
  const float* src = which ? Wk : Wq;
  if (o < C1 && c < C1) v = src[o * C1 + c];
  (which ? Wkh : Wqh)[r] = (_Float16)v;
}

// ---------------------------------------------------------------------------
// K0c: fold CWV_kk = conv_w[:,:,kk] @ Wv  (264x264 each), f16 padded to CPxCP
// ---------------------------------------------------------------------------
__global__ void k_fold_cwv(const float* __restrict__ conv_w, const float* __restrict__ Wv,
                           _Float16* __restrict__ CWVh) {
  int t = blockIdx.x * blockDim.x + threadIdx.x;     // 3*CP*CP threads
  int c = t % CP; int o = (t / CP) % CP; int kk = t / (CP * CP);
  float acc = 0.f;
  if (o < C1 && c < C1) {
    for (int m = 0; m < C1; ++m)
      acc += conv_w[(o * C1 + m) * TOPK + kk] * Wv[m * C1 + c];
  }
  CWVh[t] = (_Float16)acc;
}

// ---------------------------------------------------------------------------
// K1: build x2T = f16 unshuffled(concat(x, coords)), K-major (NB, NSP, CP)
// ---------------------------------------------------------------------------
__global__ void k_build_x2(const float* __restrict__ x, _Float16* __restrict__ x2T) {
  int cc = threadIdx.x;                              // 0..287
  int n  = blockIdx.x;                               // 0..1023
  int b  = blockIdx.y;
  float v = 0.f;
  if (cc < C1) {
    int c = cc >> 2, s1 = (cc >> 1) & 1, s2 = cc & 1;
    int h = ((n >> 5) << 1) + s1;
    int w = ((n & 31) << 1) + s2;
    if (c < CIN) {
      v = x[(((size_t)b * CIN + c) * HW + h) * HW + w];
    } else {
      float fi = (float)h, fj = (float)w;
      float nrm = sqrtf(fi * fi + fj * fj);
      float d = fmaxf(nrm, 1e-12f);
      v = (c == CIN ? fi : fj) / d;
    }
  }
  x2T[((size_t)b * NSP + n) * CP + cc] = (_Float16)v;
}

// ---------------------------------------------------------------------------
// K1b: gather sampled rows x_sT (NB, MS, CP)  (coalesced row copies)
// ---------------------------------------------------------------------------
__global__ void k_gather_xs(const _Float16* __restrict__ x2T, const int* __restrict__ flat_idx,
                            _Float16* __restrict__ xsT) {
  int cc = threadIdx.x;                              // 288
  int m = blockIdx.x; int b = blockIdx.y;
  xsT[((size_t)b * MS + m) * CP + cc] =
      x2T[((size_t)b * NSP + flat_idx[m]) * CP + cc];
}

// ---------------------------------------------------------------------------
// Generic WMMA GEMM: D(16x16 tile) = sum_k A(16x32) x B(32x16), f32 acc.
//   A: row-major [M][K] f16 (lda = K stride)
//   B: K-major   [N][K] f16 (ldb = K stride)  -> all-vector b128 loads
// Block = 128 threads = 4 waves covering 4 adjacent N tiles.
// z = bb*inner + j selects operand batch offsets.
// mode 1: f16 transposed store Ct[N][M] (one half8 store per lane)
// mode 2: f32 scaled row-major store C[M][N]
// Fragment layouts per CDNA5 ISA 7.12.2 (wave32):
//   A elem i: M = lane&15, K = (i>>3)*16 + (lane>>4)*8 + (i&7)
//   B elem i: N = lane&15, K = (i>>3)*16 + (lane>>4)*8 + (i&7)
//   D elem r: N = lane&15, M = r + 8*(lane>>4)
// ---------------------------------------------------------------------------
__global__ void k_wmma_gemm(const _Float16* __restrict__ Abase, const _Float16* __restrict__ Bbase,
                            void* __restrict__ Cbase,
                            int lda, int ldb, int ldc, int ktiles,
                            int inner, long long aJ, long long aB,
                            long long bJ, long long bB, long long cZ,
                            int mode, float scale)
{
  int z  = blockIdx.z;
  int j  = z % inner, bb = z / inner;
  const _Float16* A = Abase + (size_t)j * aJ + (size_t)bb * aB;
  const _Float16* B = Bbase + (size_t)j * bJ + (size_t)bb * bB;

  int lane = threadIdx.x & 31;
  int wv   = threadIdx.x >> 5;                   // 0..3: N tile within block
  int mo = blockIdx.y << 4;
  int no = ((blockIdx.x << 2) + wv) << 4;
  int lrow = lane & 15;          // M index (A) / N index (B,D)
  int hsel = lane >> 4;          // half-wave select

  const _Float16* arow = A + (size_t)(mo + lrow) * lda + (hsel << 3);
  const _Float16* brow = B + (size_t)(no + lrow) * ldb + (hsel << 3);

  f32x8 acc = {0.f, 0.f, 0.f, 0.f, 0.f, 0.f, 0.f, 0.f};

  for (int kt = 0; kt < ktiles; ++kt) {
    half8 alo = *(const half8*)(arow);        // K = hsel*8  + 0..7
    half8 ahi = *(const half8*)(arow + 16);   // K = 16 + hsel*8 + 0..7
    half8 blo = *(const half8*)(brow);        // K = hsel*8  + 0..7
    half8 bhi = *(const half8*)(brow + 16);   // K = 16 + hsel*8 + 0..7
    __builtin_prefetch(arow + 32, 0, 0);      // global_prefetch_b8 (next K tile)
    __builtin_prefetch(brow + 32, 0, 0);

    half16 a, bf;
#pragma unroll
    for (int i = 0; i < 8; ++i) {
      a[i] = alo[i];  a[i + 8] = ahi[i];
      bf[i] = blo[i]; bf[i + 8] = bhi[i];
    }

    acc = __builtin_amdgcn_wmma_f32_16x16x32_f16(
        /*neg_a=*/false, a, /*neg_b=*/false, bf,
        /*c_mod=*/(short)0, acc, /*reuse_a=*/false, /*reuse_b=*/false);

    arow += 32;
    brow += 32;
  }

  int col  = no + lrow;                 // N index
  int rowb = mo + (hsel << 3);          // first M index of this lane
  if (mode == 2) {
    float* C = (float*)Cbase + (size_t)z * cZ;
#pragma unroll
    for (int r = 0; r < 8; ++r) C[(size_t)(rowb + r) * ldc + col] = acc[r] * scale;
  } else {
    _Float16* C = (_Float16*)Cbase + (size_t)z * cZ;
    half8 hv;
#pragma unroll
    for (int r = 0; r < 8; ++r) hv[r] = (_Float16)acc[r];
    *(half8*)(C + (size_t)col * ldc + rowb) = hv;   // contiguous 16B store
  }
}

// ---------------------------------------------------------------------------
// K4: deterministic two-pass global max of sim  ->  big = max + 1
// ---------------------------------------------------------------------------
__global__ void k_max_partial(const float* __restrict__ sim, float* __restrict__ partials,
                              long long total) {
  __shared__ float s[256];
  long long stride = (long long)gridDim.x * blockDim.x;
  float m = -3.4e38f;
  for (long long i = (long long)blockIdx.x * blockDim.x + threadIdx.x; i < total; i += stride)
    m = fmaxf(m, sim[i]);
  s[threadIdx.x] = m; __syncthreads();
  for (int w = 128; w > 0; w >>= 1) {
    if (threadIdx.x < w) s[threadIdx.x] = fmaxf(s[threadIdx.x], s[threadIdx.x + w]);
    __syncthreads();
  }
  if (threadIdx.x == 0) partials[blockIdx.x] = s[0];
}

__global__ void k_max_final(const float* __restrict__ partials, int n, float* __restrict__ big) {
  __shared__ float s[256];
  float m = -3.4e38f;
  for (int i = threadIdx.x; i < n; i += 256) m = fmaxf(m, partials[i]);
  s[threadIdx.x] = m; __syncthreads();
  for (int w = 128; w > 0; w >>= 1) {
    if (threadIdx.x < w) s[threadIdx.x] = fmaxf(s[threadIdx.x], s[threadIdx.x + w]);
    __syncthreads();
  }
  if (threadIdx.x == 0) *big = s[0] + 1.0f;
}

// ---------------------------------------------------------------------------
// K5: per (b,n) top-3 (with diagonal override to big), softmax, gather from GT
//     out_lo[b,o,n] = conv_b[o] + sum_kk attn[kk] * GT[b,kk][idx[kk]][o]
// ---------------------------------------------------------------------------
__global__ void k_topk_out(const float* __restrict__ sim, const float* __restrict__ bigp,
                           const int* __restrict__ inv_idx, const _Float16* __restrict__ GTh,
                           const float* __restrict__ conv_b, float* __restrict__ out_lo)
{
  int n = blockIdx.x, b = blockIdx.y;
  __shared__ float s_val[MS];
  __shared__ float s_attn[TOPK];
  __shared__ int   s_idx[TOPK];

  // stage the sim row coalesced, applying the diagonal override
  {
    const float* row = sim + ((size_t)b * NSP + n) * MS;
    int movr = inv_idx[n];
    float big = *bigp;
    for (int m = threadIdx.x; m < MS; m += blockDim.x)
      s_val[m] = (m == movr) ? big : row[m];
  }
  __syncthreads();

  if (threadIdx.x == 0) {
    float v0 = -3.4e38f, v1 = v0, v2 = v0;
    int i0 = 0, i1 = 0, i2 = 0;
    for (int m = 0; m < MS; ++m) {
      float val = s_val[m];
      if (val > v0)      { v2 = v1; i2 = i1; v1 = v0; i1 = i0; v0 = val; i0 = m; }
      else if (val > v1) { v2 = v1; i2 = i1; v1 = val; i1 = m; }
      else if (val > v2) { v2 = val; i2 = m; }
    }
    float e0 = 1.0f, e1 = expf(v1 - v0), e2 = expf(v2 - v0);
    float inv = 1.0f / (e0 + e1 + e2);
    s_attn[0] = e0 * inv; s_attn[1] = e1 * inv; s_attn[2] = e2 * inv;
    s_idx[0] = i0; s_idx[1] = i1; s_idx[2] = i2;
  }
  __syncthreads();

  int o = threadIdx.x;            // blockDim = 288, active o < 264
  if (o < C1) {
    float acc = conv_b[o];
#pragma unroll
    for (int kk = 0; kk < TOPK; ++kk) {
      // GT layout [b][kk][m][o] -> coalesced over o
      acc += s_attn[kk] *
             (float)GTh[(((size_t)b * TOPK + kk) * MS + s_idx[kk]) * CP + o];
    }
    out_lo[((size_t)b * C1 + o) * NSP + n] = acc;
  }
}

// ---------------------------------------------------------------------------
// K6: pixel_shuffle(2) + pointwise 66->64 + bias
// ---------------------------------------------------------------------------
__global__ void k_shuffle_pw(const float* __restrict__ out_lo, const float* __restrict__ pw_w,
                             const float* __restrict__ pw_b, float* __restrict__ out)
{
  int t = blockIdx.x * blockDim.x + threadIdx.x;     // NB*COUT*HW*HW
  int w = t & 63, h = (t >> 6) & 63, o = (t >> 12) & 63, b = t >> 18;
  int n = ((h >> 1) << 5) + (w >> 1);
  int soff = (h & 1) * 2 + (w & 1);
  float acc = pw_b[o];
  const float* base = out_lo + (size_t)b * C1 * NSP + n;
#pragma unroll 6
  for (int c = 0; c < CPW; ++c)
    acc += pw_w[o * CPW + c] * base[(size_t)(c * 4 + soff) * NSP];
  out[t] = acc;
}

// ---------------------------------------------------------------------------
// Launcher
// ---------------------------------------------------------------------------
extern "C" void kernel_launch(void* const* d_in, const int* in_sizes, int n_in,
                              void* d_out, int out_size, void* d_ws, size_t ws_size,
                              hipStream_t stream)
{
  (void)in_sizes; (void)n_in; (void)out_size; (void)ws_size;
  const float* x      = (const float*)d_in[0];
  const float* Wq     = (const float*)d_in[1];
  const float* Wk     = (const float*)d_in[2];
  const float* Wv     = (const float*)d_in[3];
  const float* conv_w = (const float*)d_in[4];
  const float* conv_b = (const float*)d_in[5];
  const float* pw_w   = (const float*)d_in[6];
  const float* pw_b   = (const float*)d_in[7];
  float* out = (float*)d_out;

  char* ws = (char*)d_ws;
  size_t off = 0;
  auto alloc = [&](size_t bytes) -> char* {
    char* p = ws + off;
    off = (off + bytes + 255) & ~(size_t)255;
    return p;
  };
  int*      flat_idx = (int*)alloc((size_t)MS * 4);
  int*      inv_idx  = (int*)alloc((size_t)NSP * 4);
  float*    bigp     = (float*)alloc(256);
  float*    partials = (float*)alloc((size_t)4096 * 4);
  _Float16* Wqh      = (_Float16*)alloc((size_t)CP * CP * 2);
  _Float16* Wkh      = (_Float16*)alloc((size_t)CP * CP * 2);
  _Float16* CWVh     = (_Float16*)alloc((size_t)3 * CP * CP * 2);
  _Float16* x2T      = (_Float16*)alloc((size_t)NB * NSP * CP * 2);  // [b][n][c]
  _Float16* xsT      = (_Float16*)alloc((size_t)NB * MS * CP * 2);   // [b][m][c]
  _Float16* qTh      = (_Float16*)alloc((size_t)NB * NSP * CP * 2);  // [b][n][c]
  _Float16* kTh      = (_Float16*)alloc((size_t)NB * MS * CP * 2);   // [b][m][c]
  _Float16* GTh      = (_Float16*)alloc((size_t)NB * 3 * MS * CP * 2); // [b][kk][m][o]
  float*    sim      = (float*)alloc((size_t)NB * NSP * MS * 4);     // [b][n][m]
  float*    outlo    = (float*)alloc((size_t)NB * C1 * NSP * 4);     // [b][o][n]

  // Setup / conversions
  k_setup_idx<<<1, 1024, 0, stream>>>(flat_idx, inv_idx);
  k_cvt_w<<<(2 * CP * CP) / 256, 256, 0, stream>>>(Wq, Wk, Wqh, Wkh);
  k_fold_cwv<<<(3 * CP * CP) / 256, 256, 0, stream>>>(conv_w, Wv, CWVh);
  k_build_x2<<<dim3(NSP, NB), CP, 0, stream>>>(x, x2T);
  k_gather_xs<<<dim3(MS, NB), CP, 0, stream>>>(x2T, flat_idx, xsT);

  // q = Wq * x2   -> qT (f16, [n][c]): A operand of the sim GEMM
  k_wmma_gemm<<<dim3(NSP / 64, CP / 16, NB), 128, 0, stream>>>(
      Wqh, x2T, qTh, CP, CP, CP, CP / 32,
      1, 0, 0, 0, (long long)NSP * CP, (long long)NSP * CP, 1, 0.f);

  // k = Wk * x_s  -> kT (f16, [m][c]): B operand (K-major) of the sim GEMM
  k_wmma_gemm<<<dim3(MS / 64, CP / 16, NB), 128, 0, stream>>>(
      Wkh, xsT, kTh, CP, CP, CP, CP / 32,
      1, 0, 0, 0, (long long)MS * CP, (long long)MS * CP, 1, 0.f);

  // G_kk = (conv_w_kk @ Wv) * x_s  -> GT (f16, [b][kk][m][o])
  k_wmma_gemm<<<dim3(MS / 64, CP / 16, NB * 3), 128, 0, stream>>>(
      CWVh, xsT, GTh, CP, CP, CP, CP / 32,
      3, (long long)CP * CP, 0, 0, (long long)MS * CP, (long long)MS * CP, 1, 0.f);

  // sim = qT * kT^T / sqrt(C1)  (f32, [n][m])
  k_wmma_gemm<<<dim3(MS / 64, NSP / 16, NB), 128, 0, stream>>>(
      qTh, kTh, sim, CP, CP, MS, CP / 32,
      1, 0, (long long)NSP * CP, 0, (long long)MS * CP, (long long)NSP * MS,
      2, 1.0f / sqrtf((float)C1));

  // big = max(sim) + 1  (deterministic two-pass)
  k_max_partial<<<4096, 256, 0, stream>>>(sim, partials, (long long)NB * NSP * MS);
  k_max_final<<<1, 256, 0, stream>>>(partials, 4096, bigp);

  // top-3 + softmax + gather-weighted output at low res
  k_topk_out<<<dim3(NSP, NB), CP, 0, stream>>>(sim, bigp, inv_idx, GTh, conv_b, outlo);

  // pixel_shuffle + pointwise
  k_shuffle_pw<<<(NB * COUT * HW * HW) / 256, 256, 0, stream>>>(outlo, pw_w, pw_b, out);
}
